// NeuralODEFunc_25185688224022
// MI455X (gfx1250) — matrix operand order repaired and checked by reading, not compiled, not run on hardware
//
#include <hip/hip_runtime.h>

#define N_NODES 50000
#define HIDDEN  512
#define N_EDGES 160000
#define LN_EPS  1e-5f

typedef __attribute__((ext_vector_type(16))) __bf16 v16bf;
typedef __attribute__((ext_vector_type(8)))  float  v8f;

union FragBF { uint4 q[2]; v16bf v; };

__device__ __forceinline__ unsigned short f32_to_bf16_rne(float f) {
  unsigned u = __float_as_uint(f);
  u += 0x7FFFu + ((u >> 16) & 1u);           // round-to-nearest-even
  return (unsigned short)(u >> 16);
}

__device__ __forceinline__ float wave_sum(float s) {
  for (int o = 16; o > 0; o >>= 1) s += __shfl_down(s, o);
  return s;
}

// ---------------- CSR build (edges grouped by destination; kills hot-path atomics) ----

__global__ void hist_kernel(const int* __restrict__ dst, int* __restrict__ cnt) {
  int e = blockIdx.x * 256 + threadIdx.x;
  if (e < N_EDGES) atomicAdd(&cnt[dst[e]], 1);
}

// single-block exclusive scan of 50000 counts -> rowptr[0..N_NODES]
__global__ __launch_bounds__(256) void scan_kernel(const int* __restrict__ cnt,
                                                   int* __restrict__ rowptr) {
  __shared__ int part[256];
  const int tid = threadIdx.x;
  const int CH = (N_NODES + 255) / 256;   // 196
  const int base = tid * CH;
  int s = 0;
  for (int i = 0; i < CH; ++i) {
    int idx = base + i;
    if (idx < N_NODES) s += cnt[idx];
  }
  part[tid] = s;
  __syncthreads();
  for (int off = 1; off < 256; off <<= 1) {   // Hillis-Steele inclusive scan
    int v = (tid >= off) ? part[tid - off] : 0;
    __syncthreads();
    part[tid] += v;
    __syncthreads();
  }
  int run = (tid == 0) ? 0 : part[tid - 1];
  for (int i = 0; i < CH; ++i) {
    int idx = base + i;
    if (idx < N_NODES) { rowptr[idx] = run; run += cnt[idx]; }
  }
  if (tid == 255) rowptr[N_NODES] = run;
}

__global__ void dinv_kernel(const int* __restrict__ cnt, float* __restrict__ dinv) {
  int i = blockIdx.x * 256 + threadIdx.x;
  if (i < N_NODES) {
    int d = cnt[i];
    dinv[i] = (d > 0) ? rsqrtf((float)d) : 0.0f;
  }
}

__global__ void fill_kernel(const int* __restrict__ src, const int* __restrict__ dst,
                            const float* __restrict__ dinv, const int* __restrict__ rowptr,
                            int* __restrict__ cursor, int* __restrict__ esrc,
                            float* __restrict__ enrm) {
  int e = blockIdx.x * 256 + threadIdx.x;
  if (e >= N_EDGES) return;
  int d = dst[e], s = src[e];
  int pos = rowptr[d] + atomicAdd(&cursor[d], 1);
  esrc[pos] = s;
  enrm[pos] = dinv[s] * dinv[d];
}

// ---------------- weight pack: f32 W[K x 512] -> bf16 WMMA B-fragments ----------------
// Layout: [t = k/32][nt = n/16][lane 0..31][16 halves]; lane (hi=l>>4, n=nt*16+(l&15))
// holds contiguous K[t*32 + hi*16 .. +16) for its column -> GEMM reads 32B/lane b128s.
__global__ void pack_b_frag(const float* __restrict__ W, unsigned short* __restrict__ Bpk,
                            int Krows) {
  int idx = blockIdx.x * 256 + threadIdx.x;
  int lane = idx & 31;
  int nt = (idx >> 5) & 31;
  int t = idx >> 10;
  if (t >= Krows / 32) return;
  int hi = lane >> 4, l16 = lane & 15;
  int col = nt * 16 + l16;
  unsigned short* o = Bpk + (size_t)idx * 16;
  #pragma unroll
  for (int kq = 0; kq < 16; ++kq) {
    int k = t * 32 + hi * 16 + kq;
    o[kq] = f32_to_bf16_rne(W[(size_t)k * HIDDEN + col]);
  }
}

__global__ void cvt_f32_bf16(const float* __restrict__ x, unsigned short* __restrict__ y, int n) {
  int i = (blockIdx.x * 256 + threadIdx.x) * 4;
  if (i + 3 < n) {
    float4 v = *(const float4*)(x + i);
    y[i + 0] = f32_to_bf16_rne(v.x);
    y[i + 1] = f32_to_bf16_rne(v.y);
    y[i + 2] = f32_to_bf16_rne(v.z);
    y[i + 3] = f32_to_bf16_rne(v.w);
  }
}

// ---------------- WMMA bf16 GEMM (no LDS): C[M x 512] (+)= A[M x 512] * Bpk ----------
// 256 threads = 8 wave32 waves; block tile 128(M) x 64(N); each wave: 16 rows x 4 tiles.
// A/B fragment loads use running 64-bit pointers + immediate offsets so the inner loop
// is just clause'd global_load_b128s + 4 WMMAs + two pointer bumps.
__global__ __launch_bounds__(256) void gemm_bf16_wmma(
    const unsigned short* __restrict__ A,    // M x 512 bf16 row-major
    const unsigned short* __restrict__ Bpk,  // packed fragments (see pack_b_frag)
    float* __restrict__ C,                   // M x 512 f32
    int M, int accumulate)
{
  const int lane = threadIdx.x & 31;
  const int wave = threadIdx.x >> 5;
  const int hi = lane >> 4, l16 = lane & 15;
  const int mBase = blockIdx.x * 128 + wave * 16;
  const int nBase = blockIdx.y * 64;
  const int nT0 = nBase >> 4;
  const bool full = (mBase + 16 <= M);

  // A-frag row for this lane (ISA 16-bit 16x32 layout: lane m=l16); clamp keeps loads
  // in-bounds — a clamped row only feeds the matching (out-of-range) C row.
  const unsigned short* aptr =
      A + (size_t)(full ? mBase + l16 : min(mBase + l16, M - 1)) * HIDDEN + hi * 16;
  // Running B pointer: tile stride 1024B (32 lanes x 32B); per-K-step stride 32 tiles.
  const unsigned short* bptr = Bpk + ((size_t)nT0 * 32 + lane) * 16;

  v8f acc[4];
  if (accumulate) {
    for (int j = 0; j < 4; ++j)
      #pragma unroll
      for (int r = 0; r < 8; ++r) {
        int row = mBase + hi * 8 + r;
        int col = nBase + j * 16 + l16;
        acc[j][r] = (full || row < M) ? C[(size_t)row * HIDDEN + col] : 0.0f;
      }
  } else {
    v8f z = {0.f, 0.f, 0.f, 0.f, 0.f, 0.f, 0.f, 0.f};
    for (int j = 0; j < 4; ++j) acc[j] = z;
  }

  #pragma unroll 2
  for (int t = 0; t < HIDDEN / 32; ++t) {
    FragBF fa;   // contiguous K[kk+hi*16 .. +16)
    fa.q[0] = *(const uint4*)(aptr);
    fa.q[1] = *(const uint4*)(aptr + 8);
    if (t + 1 < HIDDEN / 32)
      __builtin_prefetch(aptr + 32, 0, 1);  // global_prefetch_b8 next K tile
    #pragma unroll
    for (int j = 0; j < 4; ++j) {
      FragBF fb;   // immediate offsets: j*1024B and +16B off the running pointer
      fb.q[0] = *(const uint4*)(bptr + j * 512);
      fb.q[1] = *(const uint4*)(bptr + j * 512 + 8);
      acc[j] = __builtin_amdgcn_wmma_f32_16x16x32_bf16(
          false, fa.v, false, fb.v, (short)0, acc[j], false, false);
    }
    aptr += 32;          // next K chunk of this row
    bptr += 32 * 512;    // 32 tiles * 512 halves = 32KB per K-step
  }

  // C/D layout: lane (n=l16, m-block=hi*8), VGPR r -> M = r
  if (full) {
    for (int j = 0; j < 4; ++j)
      #pragma unroll
      for (int r = 0; r < 8; ++r)
        C[(size_t)(mBase + hi * 8 + r) * HIDDEN + nBase + j * 16 + l16] = acc[j][r];
  } else {
    for (int j = 0; j < 4; ++j)
      for (int r = 0; r < 8; ++r) {
        int row = mBase + hi * 8 + r;
        if (row < M) C[(size_t)row * HIDDEN + nBase + j * 16 + l16] = acc[j][r];
      }
  }
}

// ---------------- fused gather-aggregate + bias + LayerNorm (+ bf16 mirror) ----------
// One 256-thread block per node; 2 columns per thread; no atomics anywhere.
__global__ __launch_bounds__(256) void agg_ln_kernel(
    const float* __restrict__ xw, const int* __restrict__ rowptr,
    const int* __restrict__ esrc, const float* __restrict__ enrm,
    const float* __restrict__ b, const float* __restrict__ g, const float* __restrict__ be,
    float* __restrict__ yf, unsigned short* __restrict__ ybf)
{
  const int n = blockIdx.x;
  const int tid = threadIdx.x;
  const int lane = tid & 31, wv = tid >> 5;
  const int beg = rowptr[n], end = rowptr[n + 1];

  float s0 = b[tid], s1 = b[tid + 256];
  for (int p = beg; p < end; ++p) {
    const float* xr = xw + (size_t)esrc[p] * HIDDEN;
    float w = enrm[p];
    s0 += xr[tid] * w;
    s1 += xr[tid + 256] * w;
  }

  __shared__ float red[8];
  __shared__ float stat[2];
  float s = wave_sum(s0 + s1);
  if (!lane) red[wv] = s;
  __syncthreads();
  if (tid == 0) {
    float t = 0.f;
    for (int i = 0; i < 8; ++i) t += red[i];
    stat[0] = t * (1.0f / HIDDEN);
  }
  __syncthreads();
  float mu = stat[0];
  float d0 = s0 - mu, d1 = s1 - mu;
  float q = wave_sum(d0 * d0 + d1 * d1);
  if (!lane) red[wv] = q;
  __syncthreads();
  if (tid == 0) {
    float t = 0.f;
    for (int i = 0; i < 8; ++i) t += red[i];
    stat[1] = rsqrtf(t * (1.0f / HIDDEN) + LN_EPS);
  }
  __syncthreads();
  float rs = stat[1];
  float o0 = d0 * rs * g[tid] + be[tid];
  float o1 = d1 * rs * g[tid + 256] + be[tid + 256];
  size_t base = (size_t)n * HIDDEN;
  yf[base + tid] = o0;
  yf[base + tid + 256] = o1;
  ybf[base + tid] = f32_to_bf16_rne(o0);
  ybf[base + tid + 256] = f32_to_bf16_rne(o1);
}

__global__ void gate_update(const float* __restrict__ logits, const float* __restrict__ bg,
                            const float* __restrict__ hnew, float* __restrict__ hcur,
                            unsigned short* __restrict__ hbf) {
  long long i = (long long)blockIdx.x * 256 + threadIdx.x;
  if (i >= (long long)N_NODES * HIDDEN) return;
  float gv = 1.0f / (1.0f + expf(-(logits[i] + bg[(int)(i & (HIDDEN - 1))])));
  float v = gv * hnew[i] + (1.0f - gv) * hcur[i];
  hcur[i] = v;
  hbf[i] = f32_to_bf16_rne(v);    // keep bf16 mirror current for next layer's GEMM
}

__global__ void final_kernel(const float* __restrict__ hcur, const float* __restrict__ h0,
                             const float* __restrict__ rw, float* __restrict__ out) {
  long long i = (long long)blockIdx.x * 256 + threadIdx.x;
  if (i >= (long long)N_NODES * HIDDEN) return;
  out[i] = tanhf(hcur[i]) + rw[0] * h0[i];
}

// ---------------- orchestration ----------------

static inline size_t alignup(size_t x) { return (x + 255) & ~(size_t)255; }

extern "C" void kernel_launch(void* const* d_in, const int* in_sizes, int n_in,
                              void* d_out, int out_size, void* d_ws, size_t ws_size,
                              hipStream_t stream) {
  (void)in_sizes; (void)n_in; (void)out_size; (void)ws_size;
  const float* h     = (const float*)d_in[1];
  const int*   ei    = (const int*)d_in[2];      // [2][E] int32
  const float* Wgcn  = (const float*)d_in[3];    // (3,512,512)
  const float* bgcn  = (const float*)d_in[4];    // (3,512)
  const float* gam   = (const float*)d_in[5];
  const float* bet   = (const float*)d_in[6];
  const float* Wgate = (const float*)d_in[7];    // (1024,512) K-major
  const float* bgate = (const float*)d_in[8];
  const float* rw    = (const float*)d_in[9];
  const int* src = ei;
  const int* dst = ei + N_EDGES;

  const size_t NB  = (size_t)N_NODES * HIDDEN * sizeof(float);
  const size_t NBH = (size_t)N_NODES * HIDDEN * sizeof(unsigned short);
  const size_t WPK = (size_t)HIDDEN * HIDDEN * sizeof(unsigned short);   // one packed 512x512

  char* ws = (char*)d_ws;
  size_t off = 0;
  float*          h_cur  = (float*)(ws + off);            off += alignup(NB);
  float*          buf1   = (float*)(ws + off);            off += alignup(NB);
  float*          buf2   = (float*)(ws + off);            off += alignup(NB);
  unsigned short* bfA    = (unsigned short*)(ws + off);   off += alignup(NBH);
  unsigned short* bfB    = (unsigned short*)(ws + off);   off += alignup(NBH);
  unsigned short* bpkGcn = (unsigned short*)(ws + off);   off += alignup(3 * WPK);
  unsigned short* bpkGat = (unsigned short*)(ws + off);   off += alignup(2 * WPK);
  int*            cnt    = (int*)(ws + off);              off += alignup(N_NODES * sizeof(int));
  int*            rowptr = (int*)(ws + off);              off += alignup((N_NODES + 1) * sizeof(int));
  float*          dinv   = (float*)(ws + off);            off += alignup(N_NODES * sizeof(float));
  int*            esrc   = (int*)(ws + off);              off += alignup(N_EDGES * sizeof(int));
  float*          enrm   = (float*)(ws + off);            off += alignup(N_EDGES * sizeof(float));

  const int NE = N_NODES * HIDDEN;

  hipMemcpyAsync(h_cur, h, NB, hipMemcpyDeviceToDevice, stream);
  hipMemsetAsync(cnt, 0, N_NODES * sizeof(int), stream);

  // CSR build
  hist_kernel<<<(N_EDGES + 255) / 256, 256, 0, stream>>>(dst, cnt);
  scan_kernel<<<1, 256, 0, stream>>>(cnt, rowptr);
  dinv_kernel<<<(N_NODES + 255) / 256, 256, 0, stream>>>(cnt, dinv);
  hipMemsetAsync(cnt, 0, N_NODES * sizeof(int), stream);   // reuse as fill cursor
  fill_kernel<<<(N_EDGES + 255) / 256, 256, 0, stream>>>(src, dst, dinv, rowptr, cnt, esrc, enrm);

  // pack all weights once (fuses f32->bf16 with WMMA-fragment layout)
  for (int i = 0; i < 3; ++i)
    pack_b_frag<<<(HIDDEN / 32) * 1024 / 256, 256, 0, stream>>>(
        Wgcn + (size_t)i * HIDDEN * HIDDEN, bpkGcn + (size_t)i * HIDDEN * HIDDEN, HIDDEN);
  pack_b_frag<<<(2 * HIDDEN / 32) * 1024 / 256, 256, 0, stream>>>(Wgate, bpkGat, 2 * HIDDEN);

  cvt_f32_bf16<<<(NE / 4 + 255) / 256, 256, 0, stream>>>(h, bfA, NE);

  dim3 gemmGrid((N_NODES + 127) / 128, HIDDEN / 64);
  for (int i = 0; i < 3; ++i) {
    // xw = h @ W_i
    gemm_bf16_wmma<<<gemmGrid, 256, 0, stream>>>(
        bfA, bpkGcn + (size_t)i * HIDDEN * HIDDEN, buf1, N_NODES, 0);
    // gather-aggregate + bias + LN (+ bf16 mirror); layer 0 writes straight into h_cur/bfA
    agg_ln_kernel<<<N_NODES, 256, 0, stream>>>(
        buf1, rowptr, esrc, enrm,
        bgcn + (size_t)i * HIDDEN, gam + (size_t)i * HIDDEN, bet + (size_t)i * HIDDEN,
        (i == 0) ? h_cur : buf2, (i == 0) ? bfA : bfB);
    if (i > 0) {
      // gate logits = h @ Wg[0:512] + h_new @ Wg[512:1024]
      gemm_bf16_wmma<<<gemmGrid, 256, 0, stream>>>(bfA, bpkGat, buf1, N_NODES, 0);
      gemm_bf16_wmma<<<gemmGrid, 256, 0, stream>>>(
          bfB, bpkGat + (size_t)16 * 32 * 32 * 16, buf1, N_NODES, 1);
      gate_update<<<(NE + 255) / 256, 256, 0, stream>>>(buf1, bgate, buf2, h_cur, bfA);
    }
  }

  final_kernel<<<(NE + 255) / 256, 256, 0, stream>>>(h_cur, h, rw, (float*)d_out);
}
